// SelfAttentionLayer_52278341927255
// MI455X (gfx1250) — matrix operand order repaired and checked by reading, not compiled, not run on hardware
//
#include <hip/hip_runtime.h>

// ---------------------------------------------------------------------------
// Fused self-attention for MI455X (gfx1250, wave32, WMMA).
//   q = Wq x + bq   k = Wk x + bk   v = Wv x + bv      (1x1 convs, K=256)
//   E^T = K_tile(16key x 32ch) x Q^T(32ch x 16query)   -> one v_wmma per tile
//   online softmax over keys (per-lane + shfl_xor(16) reductions)
//   out = V(ch x key) x P^T(key x query), 16 channel tiles, f32 accum
// Round-3 changes: explicit one-stage software pipelining of WMMA operand
// loads (V tiles, next-block K tiles, weight tiles) to eliminate the
// s_wait_loadcnt 0x0 stall in front of each v_wmma.
// ---------------------------------------------------------------------------

typedef __attribute__((ext_vector_type(16))) _Float16 v16h;
typedef __attribute__((ext_vector_type(8)))  _Float16 v8h;
typedef __attribute__((ext_vector_type(8)))  float    v8f;

#define DEVI static __device__ __forceinline__

constexpr int NB  = 4;     // batch
constexpr int CH  = 256;   // channels
constexpr int QKC = 32;    // q/k channels
constexpr int HW  = 4096;  // spatial positions

DEVI v8f wmma16(v16h a, v16h b, v8f c) {
  // D = A(16x32 f16) * B(32x16 f16) + C(16x16 f32)
  return __builtin_amdgcn_wmma_f32_16x16x32_f16(false, a, false, b, (short)0, c,
                                                false, false);
}

DEVI v16h combine8(v8h lo, v8h hi) {
  v16h r;
#pragma unroll
  for (int i = 0; i < 8; ++i) { r[i] = lo[i]; r[8 + i] = hi[i]; }
  return r;
}

// A-matrix tile loader (16x32 tile, contiguous k order, same as load_b16):
//   lanes 0-15 : K = 0..15 in elems 0..15 ; lanes 16-31 : K = 16..31
DEVI v16h load_a16(const _Float16* row, int hh) {
  const _Float16* p = row + hh * 16;
  v8h lo = *(const v8h*)(p);
  v8h hi = *(const v8h*)(p + 8);
  return combine8(lo, hi);
}

// B-matrix tile loader (32x16 tile, contiguous k order, same as load_a16):
//   lanes 0-15 : K = 0..15 in elems 0..15 ; lanes 16-31 : K = 16..31
DEVI v16h load_b16(const _Float16* row, int hh) {
  const _Float16* p = row + hh * 16;
  v8h lo = *(const v8h*)(p);
  v8h hi = *(const v8h*)(p + 8);
  return combine8(lo, hi);
}

// ---------------------------------------------------------------------------
// Kernel 0a: x[n][c][s] f32 -> xT[n][s][c] f16  (position-major for WMMA)
// ---------------------------------------------------------------------------
__global__ void k_xpose(const float* __restrict__ x, _Float16* __restrict__ xT) {
  int idx = blockIdx.x * blockDim.x + threadIdx.x;        // over NB*CH*HW
  int s   = idx & (HW - 1);
  int t   = idx >> 12;
  int c   = t & (CH - 1);
  int b   = t >> 8;
  if (b < NB) {
    xT[((size_t)(b * HW + s)) * CH + c] = (_Float16)x[idx];
  }
}

// ---------------------------------------------------------------------------
// Kernel 0b: one-time f32 -> f16 weight conversion (Wq, Wk: 32x256; Wv: 256x256)
// ---------------------------------------------------------------------------
__global__ void k_wcvt(const float* __restrict__ Wq, const float* __restrict__ Wk,
                       const float* __restrict__ Wv,
                       _Float16* __restrict__ wqh, _Float16* __restrict__ wkh,
                       _Float16* __restrict__ wvh) {
  int i = blockIdx.x * blockDim.x + threadIdx.x;          // over CH*CH
  if (i < QKC * CH) {
    wqh[i] = (_Float16)Wq[i];
    wkh[i] = (_Float16)Wk[i];
  }
  if (i < CH * CH) wvh[i] = (_Float16)Wv[i];
}

// ---------------------------------------------------------------------------
// Kernel 1: q/k projection.  One wave per (b, s-tile of 16).
//   D[M=s16, N=o16] = xT(16s x 32ic) * W^T(32ic x 16o), K=256 in 8 steps.
//   Software-pipelined by one K-step.
// ---------------------------------------------------------------------------
__global__ void k_qkproj(const _Float16* __restrict__ xT,
                         const _Float16* __restrict__ wqh, const float* __restrict__ bq,
                         const _Float16* __restrict__ wkh, const float* __restrict__ bk,
                         _Float16* __restrict__ qh, _Float16* __restrict__ kh) {
  const int lane = threadIdx.x & 31;
  const int wid  = blockIdx.x * (blockDim.x >> 5) + (threadIdx.x >> 5);
  const int st   = wid & 255;
  const int b    = wid >> 8;
  const int hh   = lane >> 4;
  const int lm   = lane & 15;

  const _Float16* xrow = xT + ((size_t)b * HW + st * 16 + lm) * CH; // A row (M=s)
  const _Float16* wq0 = wqh + (size_t)(lm)      * CH;               // B col o=lm
  const _Float16* wq1 = wqh + (size_t)(lm + 16) * CH;               // B col o=lm+16
  const _Float16* wk0 = wkh + (size_t)(lm)      * CH;
  const _Float16* wk1 = wkh + (size_t)(lm + 16) * CH;

  v8f qa0 = {}, qa1 = {}, ka0 = {}, ka1 = {};
  // stage 0
  v16h A  = load_a16(xrow, hh);
  v16h B0 = load_b16(wq0, hh);
  v16h B1 = load_b16(wq1, hh);
  v16h B2 = load_b16(wk0, hh);
  v16h B3 = load_b16(wk1, hh);
#pragma unroll
  for (int kk = 0; kk < 8; ++kk) {
    v16h An, B0n, B1n, B2n, B3n;
    if (kk < 7) {                       // preload next K-step before consuming
      int o = (kk + 1) * 32;
      An  = load_a16(xrow + o, hh);
      B0n = load_b16(wq0 + o, hh);
      B1n = load_b16(wq1 + o, hh);
      B2n = load_b16(wk0 + o, hh);
      B3n = load_b16(wk1 + o, hh);
    }
    qa0 = wmma16(A, B0, qa0);
    qa1 = wmma16(A, B1, qa1);
    ka0 = wmma16(A, B2, ka0);
    ka1 = wmma16(A, B3, ka1);
    A = An; B0 = B0n; B1 = B1n; B2 = B2n; B3 = B3n;
  }

  const float biasq0 = bq[lm], biasq1 = bq[lm + 16];
  const float biask0 = bk[lm], biask1 = bk[lm + 16];
#pragma unroll
  for (int r = 0; r < 8; ++r) {
    int s = st * 16 + r + 8 * hh;                 // D: M = r + 8*(lane/16)
    size_t base = ((size_t)b * HW + s) * QKC;
    qh[base + lm]      = (_Float16)(qa0[r] + biasq0);
    qh[base + lm + 16] = (_Float16)(qa1[r] + biasq1);
    kh[base + lm]      = (_Float16)(ka0[r] + biask0);
    kh[base + lm + 16] = (_Float16)(ka1[r] + biask1);
  }
}

// ---------------------------------------------------------------------------
// Kernel 2: v projection.  One wave per (b, s-tile of 16), 16 oc-tiles.
//   D[M=oc16, N=s16] = Wv(16oc x 32ic) * xT^T(32ic x 16s), K=256 in 8 steps.
//   Weight A-tiles software-pipelined by one tile.
// ---------------------------------------------------------------------------
__global__ void k_vproj(const _Float16* __restrict__ xT,
                        const _Float16* __restrict__ wvh, const float* __restrict__ bv,
                        _Float16* __restrict__ vh) {
  const int lane = threadIdx.x & 31;
  const int wid  = blockIdx.x * (blockDim.x >> 5) + (threadIdx.x >> 5);
  const int st   = wid & 255;
  const int b    = wid >> 8;
  const int hh   = lane >> 4;
  const int lm   = lane & 15;

  const _Float16* xrow = xT + ((size_t)b * HW + st * 16 + lm) * CH; // B col (N=s)
  const _Float16* wrow = wvh + (size_t)lm * CH;                     // A row base

  v8f acc[16];
#pragma unroll
  for (int i = 0; i < 16; ++i) acc[i] = (v8f){};

  v16h B = load_b16(xrow, hh);
  v16h A = load_a16(wrow, hh);                    // kk=0, ot=0
#pragma unroll 2
  for (int kk = 0; kk < 8; ++kk) {
    v16h Bn;
    if (kk < 7) Bn = load_b16(xrow + (kk + 1) * 32, hh);
#pragma unroll
    for (int ot = 0; ot < 16; ++ot) {
      v16h An;
      if (ot < 15) {
        An = load_a16(wrow + (size_t)(ot + 1) * 16 * CH + kk * 32, hh);
      } else if (kk < 7) {
        An = load_a16(wrow + (kk + 1) * 32, hh);  // ot=0 of next kk
      }
      acc[ot] = wmma16(A, B, acc[ot]);
      A = An;
    }
    B = Bn;
  }

#pragma unroll
  for (int ot = 0; ot < 16; ++ot) {
#pragma unroll
    for (int r = 0; r < 8; ++r) {
      int oc = ot * 16 + r + 8 * hh;
      int s  = st * 16 + lm;
      vh[((size_t)b * CH + oc) * HW + s] = (_Float16)(acc[ot][r] + bv[oc]);
    }
  }
}

// ---------------------------------------------------------------------------
// Kernel 3: fused flash attention.  One wave per (b, query-tile of 16).
// 64 keys per iteration: 4 E^T WMMAs + 32 AV WMMAs, one softmax update.
// K tiles for the next block are prefetched into registers during softmax;
// V tiles are pipelined one channel-tile ahead.
// ---------------------------------------------------------------------------
__global__ void k_attn(const _Float16* __restrict__ qh,
                       const _Float16* __restrict__ kh,
                       const _Float16* __restrict__ vh,
                       float* __restrict__ out) {
  const int lane = threadIdx.x & 31;
  const int wid  = blockIdx.x * (blockDim.x >> 5) + (threadIdx.x >> 5);
  const int qt   = wid & 255;
  const int b    = wid >> 8;
  const int hh   = lane >> 4;
  const int lm   = lane & 15;

  // Q as B-matrix (K=ch, N=query), loaded once.
  const _Float16* qrow = qh + ((size_t)b * HW + qt * 16 + lm) * QKC;
  const v16h Qb = load_b16(qrow, hh);

  const _Float16* kbase = kh + ((size_t)b * HW + lm) * QKC;
  const _Float16* vbase = vh + ((size_t)b * CH + lm) * HW;

  v8f acc[16];
#pragma unroll
  for (int i = 0; i < 16; ++i) acc[i] = (v8f){};
  float mrow = -3.0e38f;   // running max for query lm
  float lrow = 0.0f;       // running sum
  const v8f zero = {};

  // preload K tiles for the first 64-key block
  v16h Ka[4];
#pragma unroll
  for (int t = 0; t < 4; ++t)
    Ka[t] = load_a16(kbase + (size_t)(16 * t) * QKC, hh);

  for (int kb = 0; kb < HW; kb += 64) {
    // ---- E^T = K(16key x 32ch) x Q^T : four 16-key tiles per block ----
    v8f s[4];
#pragma unroll
    for (int t = 0; t < 4; ++t) s[t] = wmma16(Ka[t], Qb, zero);

    // ---- issue next block's K loads now; they fly during softmax ----
    if (kb + 64 < HW) {
      const _Float16* krn = kbase + (size_t)(kb + 64) * QKC;
#pragma unroll
      for (int t = 0; t < 4; ++t)
        Ka[t] = load_a16(krn + (size_t)(16 * t) * QKC, hh);
    }

    // ---- online softmax over 64 keys for query lm ----
    float bm = s[0][0];
#pragma unroll
    for (int t = 0; t < 4; ++t)
#pragma unroll
      for (int r = 0; r < 8; ++r) bm = fmaxf(bm, s[t][r]);
    bm = fmaxf(bm, __shfl_xor(bm, 16, 32));
    const float mnew = fmaxf(mrow, bm);
    const float corr = __expf(mrow - mnew);
    const bool  grew = (mnew > mrow);

    float p[4][8];
    float psum = 0.0f;
#pragma unroll
    for (int t = 0; t < 4; ++t)
#pragma unroll
      for (int r = 0; r < 8; ++r) {
        p[t][r] = __expf(s[t][r] - mnew);
        psum += p[t][r];
      }
    psum += __shfl_xor(psum, 16, 32);
    lrow = lrow * corr + psum;
    mrow = mnew;

    // ---- rescale accumulators only if any query's max moved ----
    if (__any(grew)) {
#pragma unroll
      for (int ct = 0; ct < 16; ++ct)
#pragma unroll
        for (int r = 0; r < 8; ++r) acc[ct][r] *= corr;
    }

    // ---- P^T (C-layout) -> two B-matrix operands via shfl_xor(16) ----
    v16h Pb0, Pb1;
#pragma unroll
    for (int r = 0; r < 8; ++r) {
      float o0 = __shfl_xor(p[0][r], 16, 32);  // lanes<16 get keys  8..15
      float o1 = __shfl_xor(p[1][r], 16, 32);  // lanes>=16 get keys 16..23
      float o2 = __shfl_xor(p[2][r], 16, 32);  // lanes<16 get keys 40..47
      float o3 = __shfl_xor(p[3][r], 16, 32);  // lanes>=16 get keys 48..55
      Pb0[r]     = (_Float16)(hh ? o1 : p[0][r]);   // elems 0..7
      Pb0[8 + r] = (_Float16)(hh ? p[1][r] : o0);   // elems 8..15
      Pb1[r]     = (_Float16)(hh ? o3 : p[2][r]);
      Pb1[8 + r] = (_Float16)(hh ? p[3][r] : o2);
    }

    // ---- out[ch, query] += V(ch x 64key) x P^T(64key x query) ----
    // pipelined one channel-tile ahead
    v16h va = load_a16(vbase + kb, hh);
    v16h vb = load_a16(vbase + kb + 32, hh);
#pragma unroll
    for (int ct = 0; ct < 16; ++ct) {
      v16h na, nb;
      if (ct < 15) {
        const _Float16* vr = vbase + (size_t)(ct + 1) * 16 * HW + kb;
        na = load_a16(vr, hh);
        nb = load_a16(vr + 32, hh);
      }
      acc[ct] = wmma16(va, Pb0, acc[ct]);
      acc[ct] = wmma16(vb, Pb1, acc[ct]);
      va = na; vb = nb;
    }
  }

  const float inv = 1.0f / lrow;
#pragma unroll
  for (int ct = 0; ct < 16; ++ct) {
#pragma unroll
    for (int r = 0; r < 8; ++r) {
      int c = ct * 16 + r + 8 * hh;
      int q = qt * 16 + lm;
      out[((size_t)b * CH + c) * HW + q] = acc[ct][r] * inv;
    }
  }
}

// ---------------------------------------------------------------------------
extern "C" void kernel_launch(void* const* d_in, const int* in_sizes, int n_in,
                              void* d_out, int out_size, void* d_ws, size_t ws_size,
                              hipStream_t stream) {
  (void)in_sizes; (void)n_in; (void)out_size; (void)ws_size;
  const float* x  = (const float*)d_in[0];
  const float* Wq = (const float*)d_in[1];
  const float* bq = (const float*)d_in[2];
  const float* Wk = (const float*)d_in[3];
  const float* bk = (const float*)d_in[4];
  const float* Wv = (const float*)d_in[5];
  const float* bv = (const float*)d_in[6];
  float* out = (float*)d_out;

  char* ws = (char*)d_ws;
  const size_t MB = 1024 * 1024;
  _Float16* xT  = (_Float16*)(ws);                  // 8 MiB  [n][s][256]
  _Float16* qh  = (_Float16*)(ws + 8  * MB);        // 1 MiB  [n][s][32]
  _Float16* kh  = (_Float16*)(ws + 9  * MB);        // 1 MiB  [n][s][32]
  _Float16* vh  = (_Float16*)(ws + 10 * MB);        // 8 MiB  [n][c][s]
  _Float16* wqh = (_Float16*)(ws + 18 * MB);        // 16 KiB [32][256]
  _Float16* wkh = (_Float16*)(ws + 18 * MB + 64 * 1024);
  _Float16* wvh = (_Float16*)(ws + 18 * MB + 128 * 1024); // 128 KiB [256][256]

  k_xpose <<<(NB * CH * HW) / 256, 256, 0, stream>>>(x, xT);
  k_wcvt  <<<(CH * CH) / 256, 256, 0, stream>>>(Wq, Wk, Wv, wqh, wkh, wvh);
  k_qkproj<<<256, 128, 0, stream>>>(xT, wqh, bq, wkh, bk, qh, kh);
  k_vproj <<<256, 128, 0, stream>>>(xT, wvh, bv, vh);
  k_attn  <<<256, 128, 0, stream>>>(qh, kh, vh, out);
}
